// Bilstm_CRF_22582938042909
// MI455X (gfx1250) — matrix operand order repaired and checked by reading, not compiled, not run on hardware
//
#include <hip/hip_runtime.h>
#include <hip/hip_bf16.h>
#include <stdint.h>

// ---------------------------------------------------------------------------
// BiLSTM-CRF forward NLL for MI455X (gfx1250, wave32, WMMA).
//   B=64, T=512, E=512, H=512, 4H=2048, L=32, V=50000
// Strategy:
//   * bf16 WMMA (v_wmma_f32_16x16x32_bf16) for all GEMMs, f32 accumulate.
//   * All tile loops compile-time; no divergent guards around WMMA (EXEC
//     must be all-1s per ISA 7.12); wave ids via readfirstlane -> SALU tiles.
//   * Input-to-gate projections precomputed as parallel GEMMs, stored
//     TIME-MAJOR ([t][b][4H]) with non-temporal stores so each scan step
//     reads one contiguous 512KB slab (streamed, NT loads).
//   * Recurrent scan: 8 WGs x 4 waves per direction; Whh slice becomes fully
//     VGPR-resident (compiler hoists B tiles via VGPR-MSB banks); cell state
//     in VGPRs for all 512 steps; per-step cross-WG arrive/spin barrier.
//   * CRF forward algorithm in one block with LDS alpha double-buffer.
// ---------------------------------------------------------------------------

typedef __attribute__((ext_vector_type(16))) __bf16 bf16x16;
typedef __attribute__((ext_vector_type(8)))  __bf16 bf16x8;
typedef __attribute__((ext_vector_type(8)))  float  f32x8;

#define BT     32768      // B*T
#define HDIM   512
#define G4     2048       // 4H
#define NLAB   32
#define TT     512
#define NB     64

__device__ __forceinline__ uint16_t f2bf(float f) {
  uint32_t u = __float_as_uint(f);
  uint32_t r = u + 0x7FFFu + ((u >> 16) & 1u);   // round-to-nearest-even
  return (uint16_t)(r >> 16);
}

__device__ __forceinline__ f32x8 wmma_bf16(bf16x16 a, bf16x16 b, f32x8 c) {
  return __builtin_amdgcn_wmma_f32_16x16x32_bf16(
      /*neg_a=*/false, a, /*neg_b=*/false, b,
      /*c_mod=*/(short)0, c, /*reuse_a=*/false, /*reuse_b=*/false);
}

// A tile 16(M)x32(K) bf16 from row-major [.. , lda] activations.
// ISA layout: lanes 0-15 -> M=lane, halves 0..7=K(base+0..7), 8..15=K(base+16..23)
//             lanes16-31 -> M=lane-16, halves 0..7=K(base+8..15), 8..15=K(base+24..31)
__device__ __forceinline__ bf16x16 load_a_tile(const uint16_t* __restrict__ A,
                                               int lda, int mbase, int kbase, int lane) {
  int m = mbase + (lane & 15);
  int k = kbase + ((lane >> 4) << 3);     // 0 or 8
  const __bf16* row = (const __bf16*)(A + (size_t)m * lda + k);
  union { bf16x16 v; struct { bf16x8 lo; bf16x8 hi; } s; } u;
  u.s.lo = *(const bf16x8*)(row);
  u.s.hi = *(const bf16x8*)(row + 16);
  return u.v;
}

// Pre-packed B tile 32(K)x16(N): per lane 16 contiguous halves (32B).
// Pack convention: lane%16 = N-local, half s -> K-local (s&8)*2 + 8*(lane/16) + (s&7).
__device__ __forceinline__ bf16x16 load_b_tile(const uint16_t* __restrict__ Bp,
                                               int KT, int kt, int nt, int lane) {
  const __bf16* p = (const __bf16*)(Bp + (((size_t)nt * KT + kt) * 32 + lane) * 16);
  return *(const bf16x16*)p;
}

// ------------------------------ prep kernels -------------------------------

__global__ void embed_cast_k(const int* __restrict__ tok, const float* __restrict__ emb,
                             uint16_t* __restrict__ xe, long total) {
  long i = (long)blockIdx.x * blockDim.x + threadIdx.x;
  if (i >= total) return;
  long row = i >> 9;              // /512
  int  e   = (int)(i & 511);
  int  t   = tok[row];
  xe[i] = f2bf(emb[(size_t)t * 512 + e]);
}

// Pack weight into WMMA-B tile layout. transposed=1: W is [N][K]; else [K][N].
__global__ void pack_b_k(const float* __restrict__ W, uint16_t* __restrict__ Bp,
                         int K, int N, int transposed, long total) {
  long idx = (long)blockIdx.x * blockDim.x + threadIdx.x;
  if (idx >= total) return;
  int  h    = (int)(idx & 15);
  int  lane = (int)((idx >> 4) & 31);
  long rest = idx >> 9;
  int  KT   = K >> 5;
  int  kt   = (int)(rest % KT);
  int  nt   = (int)(rest / KT);
  int  k    = (kt << 5) + ((h & 8) << 1) + ((lane >> 4) << 3) + (h & 7);
  int  n    = (nt << 4) + (lane & 15);
  float v = transposed ? W[(size_t)n * K + k] : W[(size_t)k * N + n];
  Bp[idx] = f2bf(v);
}

__global__ void bias_sum_k(const float* a, const float* b, float* o, int n) {
  int i = blockIdx.x * blockDim.x + threadIdx.x;
  if (i < n) o[i] = a[i] + b[i];
}

__global__ void scan_init_k(uint16_t* hbuf, unsigned* ctr, int n) {
  int i = blockIdx.x * blockDim.x + threadIdx.x;
  if (i < n) hbuf[i] = 0;
  if (i < 2) ctr[i] = 0;
}

// ------------------------------- GEMM kernel -------------------------------
// C[M,N](f32) = A[M,K](bf16, row-major) x Bp(packed bf16) + bias[N]
// Each wave computes MW x NW 16x16 tiles; grid sized so every wave is full.
// SWZ: remap output row m = b*TT+t  ->  t*NB+b (time-major xg) + NT stores.
template <int MW, int NW, bool SWZ>
__global__ __launch_bounds__(256) void gemm_bf16_k(
    const uint16_t* __restrict__ A, const uint16_t* __restrict__ Bp,
    const float* __restrict__ bias, float* __restrict__ C,
    int M, int K, int N) {
  const int KT = K >> 5;
  const int NG = (N >> 4) / NW;
  const int wid  = __builtin_amdgcn_readfirstlane(threadIdx.x >> 5);
  const int lane = threadIdx.x & 31;
  const int gwid = blockIdx.x * 8 + wid;
  const int mg = gwid / NG;
  const int ng = gwid - mg * NG;
  const int mbase = mg * (MW * 16);
  const int ntb   = ng * NW;            // first N tile index

  f32x8 acc[MW][NW] = {};
  for (int kt = 0; kt < KT; ++kt) {
    bf16x16 a[MW];
#pragma unroll
    for (int i = 0; i < MW; ++i)
      a[i] = load_a_tile(A, K, mbase + 16 * i, kt << 5, lane);
    if (kt + 1 < KT)   // uniform branch: prefetch next K-step B tiles
      __builtin_prefetch((const void*)(Bp + (((size_t)ntb * KT + kt + 1) * 32 + lane) * 16), 0, 1);
#pragma unroll
    for (int q = 0; q < NW; ++q) {
      bf16x16 b = load_b_tile(Bp, KT, kt, ntb + q, lane);
#pragma unroll
      for (int i = 0; i < MW; ++i)
        acc[i][q] = wmma_bf16(a[i], b, acc[i][q]);
    }
  }
  const int nl  = lane & 15;
  const int mro = (lane >> 4) << 3;     // 0 or 8
#pragma unroll
  for (int q = 0; q < NW; ++q) {
    int n = ((ntb + q) << 4) + nl;
    float bz = bias[n];
#pragma unroll
    for (int i = 0; i < MW; ++i) {
#pragma unroll
      for (int r = 0; r < 8; ++r) {
        int m = mbase + 16 * i + mro + r;          // m = b*TT + t
        float v = acc[i][q][r] + bz;
        if (SWZ) {
          int mrow = ((m & (TT - 1)) << 6) + (m >> 9);   // t*NB + b
          __builtin_nontemporal_store(v, &C[(size_t)mrow * N + n]);
        } else {
          C[(size_t)m * N + n] = v;
        }
      }
    }
  }
}

// ------------------------------- LSTM scan ---------------------------------
// grid = 2 dirs x NWG; block = 128 (4 waves). Each wave: 16 hidden units x
// all 64 batch rows; c kept in VGPRs across all 512 steps. xg is time-major.
#define NWG 8
__global__ __launch_bounds__(128) void lstm_scan_k(
    const float* __restrict__ xg0, const float* __restrict__ xg1,
    const uint16_t* __restrict__ whh0, const uint16_t* __restrict__ whh1,
    uint16_t* __restrict__ hbuf,       // [2 dirs][2][64*512] bf16
    uint16_t* __restrict__ hout,       // [B*T][1024] bf16
    unsigned* __restrict__ ctr) {      // [2]
  const int dir  = __builtin_amdgcn_readfirstlane(blockIdx.x) / NWG;
  const int wg   = __builtin_amdgcn_readfirstlane(blockIdx.x) % NWG;
  const int wid  = __builtin_amdgcn_readfirstlane(threadIdx.x >> 5);
  const int lane = threadIdx.x & 31;
  const float*    xg  = dir ? xg1 : xg0;
  const uint16_t* whh = dir ? whh1 : whh0;
  uint16_t* hb = hbuf + (size_t)dir * 2 * NB * HDIM;
  unsigned* cc = ctr + dir;

  const int ub   = (wg * 4 + wid) << 4;   // unit base: 0..511 step 16 (scalar)
  const int nl   = lane & 15;
  const int mro  = (lane >> 4) << 3;
  const int unit = ub + nl;
  const int KT   = HDIM / 32;             // 16

  float creg[4][8];
#pragma unroll
  for (int a = 0; a < 4; ++a)
#pragma unroll
    for (int r = 0; r < 8; ++r) creg[a][r] = 0.f;

  for (int step = 0; step < TT; ++step) {
    int t = dir ? (TT - 1 - step) : step;
    const uint16_t* hcur = hb + (size_t)(step & 1) * NB * HDIM;
    uint16_t*       hnxt = hb + (size_t)((step + 1) & 1) * NB * HDIM;
    const float*    xgt  = xg + (size_t)t * NB * G4;   // contiguous 512KB slab

    for (int mt = 0; mt < 4; ++mt) {
      f32x8 acc[4] = {};
      for (int kt = 0; kt < KT; ++kt) {
        bf16x16 a = load_a_tile(hcur, HDIM, mt << 4, kt << 5, lane);
#pragma unroll
        for (int g = 0; g < 4; ++g) {
          int nt = (g << 5) + (ub >> 4);   // gate g, unit tile (scalar)
          bf16x16 b = load_b_tile(whh, KT, kt, nt, lane);
          acc[g] = wmma_bf16(a, b, acc[g]);
        }
      }
#pragma unroll
      for (int r = 0; r < 8; ++r) {
        int bidx = (mt << 4) + mro + r;    // batch row
        const float* xrow = xgt + (size_t)bidx * G4;
        float gi = acc[0][r] + __builtin_nontemporal_load(&xrow[0 * HDIM + unit]);
        float gf = acc[1][r] + __builtin_nontemporal_load(&xrow[1 * HDIM + unit]);
        float gg = acc[2][r] + __builtin_nontemporal_load(&xrow[2 * HDIM + unit]);
        float go = acc[3][r] + __builtin_nontemporal_load(&xrow[3 * HDIM + unit]);
        float ii = 1.f / (1.f + __expf(-gi));
        float ff = 1.f / (1.f + __expf(-gf));
        float oo = 1.f / (1.f + __expf(-go));
        float cv = ff * creg[mt][r] + ii * tanhf(gg);
        creg[mt][r] = cv;
        uint16_t hb16 = f2bf(oo * tanhf(cv));
        hnxt[(size_t)bidx * HDIM + unit] = hb16;
        hout[((size_t)bidx * TT + t) * 1024 + dir * HDIM + unit] = hb16;
      }
    }
    // cross-WG barrier for this direction (device-scope arrive + spin)
    __threadfence();
    __syncthreads();
    if (threadIdx.x == 0) {
      atomicAdd(cc, 1u);
      unsigned target = (unsigned)(step + 1) * NWG;
      while (atomicAdd(cc, 0u) < target) __builtin_amdgcn_s_sleep(1);
    }
    __syncthreads();
    __threadfence();
  }
}

// --------------------------------- CRF -------------------------------------
__global__ __launch_bounds__(1024) void crf_nll_k(
    const float* __restrict__ emis, const int* __restrict__ tags,
    const float* __restrict__ start, const float* __restrict__ endp,
    const float* __restrict__ trans, float* __restrict__ out) {
  __shared__ float sTrans[NLAB * NLAB];
  __shared__ float sAlpha[2][NB][NLAB];
  __shared__ float sNum[NB];
  __shared__ float sRes[NB];
  int tid = threadIdx.x;
  sTrans[tid] = trans[tid];                      // 1024 == 32*32

  if (tid < NB) {                                // numerator
    int b = tid;
    const int* tg = tags + (size_t)b * TT;
    float s = start[tg[0]];
    for (int t = 0; t < TT; ++t) {
      int y = tg[t];
      s += emis[((size_t)b * TT + t) * NLAB + y];
      if (t < TT - 1) s += trans[y * NLAB + tg[t + 1]];
    }
    s += endp[tg[TT - 1]];
    sNum[b] = s;
  }
  for (int p = tid; p < NB * NLAB; p += 1024) {  // alpha init
    int b = p >> 5, j = p & 31;
    sAlpha[0][b][j] = start[j] + emis[((size_t)b * TT) * NLAB + j];
  }
  __syncthreads();

  for (int t = 1; t < TT; ++t) {
    int src = (t - 1) & 1, dst = t & 1;
    for (int p = tid; p < NB * NLAB; p += 1024) {
      int b = p >> 5, j = p & 31;
      float mx = -3.4e38f;
#pragma unroll
      for (int i = 0; i < NLAB; ++i)
        mx = fmaxf(mx, sAlpha[src][b][i] + sTrans[i * NLAB + j]);
      float sum = 0.f;
#pragma unroll
      for (int i = 0; i < NLAB; ++i)
        sum += __expf(sAlpha[src][b][i] + sTrans[i * NLAB + j] - mx);
      sAlpha[dst][b][j] = mx + __logf(sum) + emis[((size_t)b * TT + t) * NLAB + j];
    }
    __syncthreads();
  }

  if (tid < NB) {
    int b = tid;
    float mx = -3.4e38f;
    for (int j = 0; j < NLAB; ++j)
      mx = fmaxf(mx, sAlpha[(TT - 1) & 1][b][j] + endp[j]);
    float sum = 0.f;
    for (int j = 0; j < NLAB; ++j)
      sum += __expf(sAlpha[(TT - 1) & 1][b][j] + endp[j] - mx);
    sRes[b] = sNum[b] - (mx + __logf(sum));
  }
  __syncthreads();
  if (tid == 0) {
    float tot = 0.f;
    for (int b = 0; b < NB; ++b) tot += sRes[b];
    out[0] = -tot;
  }
}

// ------------------------------ host driver --------------------------------
extern "C" void kernel_launch(void* const* d_in, const int* in_sizes, int n_in,
                              void* d_out, int out_size, void* d_ws, size_t ws_size,
                              hipStream_t stream) {
  const int*   datas   = (const int*)d_in[0];
  const int*   labels  = (const int*)d_in[1];
  const float* emb     = (const float*)d_in[2];
  const float* w_ih0   = (const float*)d_in[3];   // [2,2048,512]
  const float* w_ih1   = (const float*)d_in[4];   // [2,2048,1024]
  const float* w_hh    = (const float*)d_in[5];   // [2,2,2048,512]
  const float* b_ih    = (const float*)d_in[6];   // [2,2,2048]
  const float* b_hh    = (const float*)d_in[7];
  const float* w_out   = (const float*)d_in[8];   // [1024,32]
  const float* b_out   = (const float*)d_in[9];
  const float* c_start = (const float*)d_in[10];
  const float* c_end   = (const float*)d_in[11];
  const float* c_trans = (const float*)d_in[12];
  float* out = (float*)d_out;

  char* p = (char*)d_ws;
  auto carve = [&](size_t bytes) -> char* {
    char* r = p;
    p += (bytes + 255) & ~(size_t)255;
    return r;
  };
  uint16_t* xe    = (uint16_t*)carve((size_t)BT * HDIM * 2);
  uint16_t* h1    = (uint16_t*)carve((size_t)BT * 1024 * 2);
  uint16_t* h2    = (uint16_t*)carve((size_t)BT * 1024 * 2);
  float*    xg0   = (float*)carve((size_t)BT * G4 * 4);
  float*    xg1   = (float*)carve((size_t)BT * G4 * 4);
  uint16_t* wih0p = (uint16_t*)carve((size_t)2 * G4 * 512 * 2);
  uint16_t* wih1p = (uint16_t*)carve((size_t)2 * G4 * 1024 * 2);
  uint16_t* whhp  = (uint16_t*)carve((size_t)4 * G4 * 512 * 2);
  uint16_t* woutp = (uint16_t*)carve((size_t)1024 * NLAB * 2);
  float*    bsum  = (float*)carve((size_t)4 * G4 * 4);
  float*    emis  = (float*)carve((size_t)BT * NLAB * 4);
  uint16_t* hbuf  = (uint16_t*)carve((size_t)2 * 2 * NB * HDIM * 2);
  unsigned* ctr   = (unsigned*)carve(256);

  // 1. embedding gather + bf16 cast
  {
    long total = (long)BT * HDIM;
    embed_cast_k<<<(unsigned)((total + 255) / 256), 256, 0, stream>>>(datas, emb, xe, total);
  }
  // 2. weight packing + bias sums
  auto pack = [&](const float* W, uint16_t* dst, int K, int N, int tr) {
    long total = (long)K * N;
    pack_b_k<<<(unsigned)((total + 255) / 256), 256, 0, stream>>>(W, dst, K, N, tr, total);
  };
  pack(w_ih0,                       wih0p,                       512, G4, 1);
  pack(w_ih0 + (size_t)G4 * 512,    wih0p + (size_t)G4 * 512,    512, G4, 1);
  pack(w_ih1,                       wih1p,                       1024, G4, 1);
  pack(w_ih1 + (size_t)G4 * 1024,   wih1p + (size_t)G4 * 1024,   1024, G4, 1);
  for (int i = 0; i < 4; ++i)
    pack(w_hh + (size_t)i * G4 * 512, whhp + (size_t)i * G4 * 512, 512, G4, 1);
  pack(w_out, woutp, 1024, NLAB, 0);
  {
    int n = 4 * G4;
    bias_sum_k<<<(n + 255) / 256, 256, 0, stream>>>(b_ih, b_hh, bsum, n);
  }
  // big GEMMs (time-major NT output): MW=2,NW=4 -> 32768 waves -> 4096 WGs
  auto gemm_xg = [&](const uint16_t* A, const uint16_t* Bp, const float* bias,
                     float* C, int M, int K, int N) {
    int waves = (M / 32) * (N / 64);
    gemm_bf16_k<2, 4, true><<<waves / 8, 256, 0, stream>>>(A, Bp, bias, C, M, K, N);
  };
  // 3. layer-1 input projections, scan
  gemm_xg(xe, wih0p,                    bsum,      xg0, BT, 512, G4);
  gemm_xg(xe, wih0p + (size_t)G4 * 512, bsum + G4, xg1, BT, 512, G4);
  {
    int n = 2 * 2 * NB * HDIM;
    scan_init_k<<<(n + 255) / 256, 256, 0, stream>>>(hbuf, ctr, n);
  }
  lstm_scan_k<<<2 * NWG, 128, 0, stream>>>(xg0, xg1, whhp, whhp + (size_t)G4 * 512,
                                           hbuf, h1, ctr);
  // 4. layer-2 input projections, scan
  gemm_xg(h1, wih1p,                     bsum + 2 * G4, xg0, BT, 1024, G4);
  gemm_xg(h1, wih1p + (size_t)G4 * 1024, bsum + 3 * G4, xg1, BT, 1024, G4);
  {
    int n = 2 * 2 * NB * HDIM;
    scan_init_k<<<(n + 255) / 256, 256, 0, stream>>>(hbuf, ctr, n);
  }
  lstm_scan_k<<<2 * NWG, 128, 0, stream>>>(xg0, xg1,
                                           whhp + (size_t)2 * G4 * 512,
                                           whhp + (size_t)3 * G4 * 512,
                                           hbuf, h2, ctr);
  // 5. emissions: MW=2,NW=2, row-major output -> 1024 waves -> 128 WGs
  {
    int waves = (BT / 32) * (NLAB / 32);
    gemm_bf16_k<2, 2, false><<<waves / 8, 256, 0, stream>>>(h2, woutp, b_out, emis,
                                                            BT, 1024, NLAB);
  }
  // 6. CRF NLL
  crf_nll_k<<<1, 1024, 0, stream>>>(emis, labels, c_start, c_end, c_trans, out);
}
